// Multi_LSTMs_5660766896170
// MI455X (gfx1250) — matrix operand order repaired and checked by reading, not compiled, not run on hardware
//
#include <hip/hip_runtime.h>
#include <stdint.h>

// ---------------- problem constants ----------------
#define BS   8
#define NT   128
#define CH   1024
#define KH   4            // heads
#define H    512
#define G    2048         // 4*H gate dim
#define DIR  2
#define NTB  (NT*BS)      // 1024 GEMM rows
#define NREC (DIR*KH)     // 8 independent recurrences
#define WGR  16           // workgroups per recurrence
#define HSL  32           // h-columns per recurrence-WG (16*32 = 512)

typedef __attribute__((ext_vector_type(16))) __bf16 v16bf;
typedef __attribute__((ext_vector_type(8)))  float  v8f;

struct alignas(16) U4 { unsigned x, y, z, w; };
union Frag16 { U4 u[2]; v16bf v; };   // 32B: one WMMA bf16 A/B fragment per lane

__device__ __forceinline__ unsigned short f2bf(float f) {
    union { float f; unsigned u; } cv; cv.f = f;
    unsigned u = cv.u;
    return (unsigned short)((u + 0x7FFFu + ((u >> 16) & 1u)) >> 16); // RNE
}
__device__ __forceinline__ float sigm(float x) { return 1.0f / (1.0f + __expf(-x)); }

// ---------------- workspace layout (bytes) ----------------
#define OFF_X    ((size_t)0)                         // bf16 x   [KH][NTB][CH]      8 MiB
#define OFF_WIH  (OFF_X    + (size_t)KH*NTB*CH*2)    // bf16 Wih [NREC][G][CH]     32 MiB
#define OFF_WHH  (OFF_WIH  + (size_t)NREC*G*CH*2)    // bf16 Whh [NREC][G][H]      16 MiB
#define OFF_BIAS (OFF_WHH  + (size_t)NREC*G*H*2)     // f32 bias [NREC][G]         64 KiB
#define OFF_APRE (OFF_BIAS + (size_t)NREC*G*4)       // f32 Apre [NREC][NT][BS][G] 64 MiB
#define OFF_HBUF (OFF_APRE + (size_t)NREC*NTB*G*4)   // bf16 h   [NREC][2][16][H] 256 KiB
#define OFF_BARS (OFF_HBUF + (size_t)NREC*2*16*H*2)  // u32 bars [NREC][32]         1 KiB

// ---------------- prep kernels ----------------
__global__ __launch_bounds__(256) void k_cvt(const float* __restrict__ s,
                                             unsigned short* __restrict__ d, size_t n) {
    for (size_t i = (size_t)blockIdx.x * blockDim.x + threadIdx.x; i < n;
         i += (size_t)gridDim.x * blockDim.x)
        d[i] = f2bf(s[i]);
}

// x [b][t][c][k] f32 -> xbf [k][t*8+b][c] bf16
__global__ __launch_bounds__(256) void k_cvt_x(const float* __restrict__ x,
                                               unsigned short* __restrict__ d) {
    size_t n = (size_t)KH * NTB * CH;
    for (size_t i = (size_t)blockIdx.x * blockDim.x + threadIdx.x; i < n;
         i += (size_t)gridDim.x * blockDim.x) {
        int k = (int)(i / ((size_t)NTB * CH));
        size_t r = i % ((size_t)NTB * CH);
        int m = (int)(r / CH), c = (int)(r % CH);
        int t = m >> 3, b = m & 7;
        d[i] = f2bf(x[(((size_t)b * NT + t) * CH + c) * KH + k]);
    }
}

// bias = bih+bhh (both dirs), zero h buffers, zero barriers
__global__ __launch_bounds__(256) void k_prep(const float* bif, const float* bhf,
                                              const float* bib, const float* bhb,
                                              float* bias, unsigned short* hbuf,
                                              unsigned* bars) {
    const size_t NB = (size_t)NREC * G, NHB = (size_t)NREC * 2 * 16 * H, NBR = NREC * 32;
    size_t n = NB + NHB + NBR;
    for (size_t i = (size_t)blockIdx.x * blockDim.x + threadIdx.x; i < n;
         i += (size_t)gridDim.x * blockDim.x) {
        if (i < NB) {
            int d = (int)(i / ((size_t)KH * G));
            size_t loc = i - (size_t)d * KH * G;
            bias[i] = (d ? bib[loc] : bif[loc]) + (d ? bhb[loc] : bhf[loc]);
        } else if (i < NB + NHB) {
            hbuf[i - NB] = 0;
        } else {
            bars[i - NB - NHB] = 0u;
        }
    }
}

// ---------------- input-projection GEMM (batched over NREC) ----------------
// Apre[dk][m][n] = sum_c xbf[k][m][c] * wih[dk][n][c] + bias[dk][n]
// block = 128 thr (4 waves), tile 64x64, K-step 32.
__global__ __launch_bounds__(128) void k_gemm_in(const unsigned short* __restrict__ xbf,
                                                 const unsigned short* __restrict__ wih,
                                                 const float* __restrict__ bias,
                                                 float* __restrict__ apre) {
    const int dk = blockIdx.z, kk = dk & 3;
    const int nb = blockIdx.x * 64, mb = blockIdx.y * 64;
    const int w = threadIdx.x >> 5, lane = threadIdx.x & 31;
    const int m = mb + w * 16 + (lane & 15);
    const int koA = (lane >> 4) * 8;    // A k-offset per lane half
    const int koB = (lane >> 4) * 8;    // B k-offset per lane half
    const unsigned short* ap = xbf + ((size_t)kk * NTB + m) * CH;

    v8f acc[4] = {};
    for (int ks = 0; ks < CH; ks += 32) {
        Frag16 A;
        A.u[0] = *(const U4*)(ap + ks + koA);
        A.u[1] = *(const U4*)(ap + ks + koA + 16);
#pragma unroll
        for (int j = 0; j < 4; ++j) {
            int n = nb + j * 16 + (lane & 15);
            const unsigned short* bp = wih + ((size_t)dk * G + n) * CH + ks + koB;
            Frag16 B;
            B.u[0] = *(const U4*)(bp);
            B.u[1] = *(const U4*)(bp + 16);
            acc[j] = __builtin_amdgcn_wmma_f32_16x16x32_bf16(
                false, A.v, false, B.v, (short)0, acc[j], false, false);
        }
    }
#pragma unroll
    for (int j = 0; j < 4; ++j) {
        int n = nb + j * 16 + (lane & 15);
        float bv = bias[(size_t)dk * G + n];
#pragma unroll
        for (int v = 0; v < 8; ++v) {
            int mr = mb + w * 16 + v + (lane >> 4) * 8;
            apre[((size_t)dk * NTB + mr) * G + n] = acc[j][v] + bv;
        }
    }
}

// ---------------- persistent recurrent kernel ----------------
__device__ __forceinline__ void rec_barrier(unsigned* bar, unsigned target) {
    __threadfence();                      // release my h stores to device scope
    __syncthreads();
    if (threadIdx.x == 0) {
        unsigned old = __hip_atomic_fetch_add(&bar[0], 1u, __ATOMIC_ACQ_REL,
                                              __HIP_MEMORY_SCOPE_AGENT);
        if (old == WGR - 1) {
            __hip_atomic_store(&bar[0], 0u, __ATOMIC_RELAXED, __HIP_MEMORY_SCOPE_AGENT);
            __hip_atomic_fetch_add(&bar[1], 1u, __ATOMIC_RELEASE, __HIP_MEMORY_SCOPE_AGENT);
        }
        while (__hip_atomic_load(&bar[1], __ATOMIC_ACQUIRE, __HIP_MEMORY_SCOPE_AGENT) < target)
            __builtin_amdgcn_s_sleep(8);
    }
    __syncthreads();
    __threadfence();                      // acquire: invalidate stale h lines
}

// grid = NREC*WGR blocks, 256 threads (8 waves). Each WG owns 32 h-columns.
// Wave w: gate section sec=w&3 (i/f/g/o), column sub-tile half=w>>2.
__global__ __launch_bounds__(256) void k_lstm_rec(const unsigned short* __restrict__ whh,
                                                  const float* __restrict__ apre,
                                                  unsigned short* __restrict__ hbuf,
                                                  float* __restrict__ out,
                                                  unsigned* __restrict__ bars) {
    __shared__ float lds_g[4][HSL][BS];   // [section][local col][b]

    const int rec = blockIdx.x >> 4, wg = blockIdx.x & 15;
    const int dd = rec >> 2, kk = rec & 3;
    const int w = threadIdx.x >> 5, lane = threadIdx.x & 31;
    const int sec = w & 3, half = w >> 2;
    const int slice0 = wg * HSL;
    const int n = sec * H + slice0 + half * 16 + (lane & 15); // my gate column
    const unsigned short* Bbase = whh + ((size_t)rec * G + n) * H + (lane >> 4) * 8;
    unsigned short* hb = hbuf + (size_t)rec * 2 * 16 * H;
    unsigned* bar = bars + rec * 32;
    const int mrow = lane & 15, koA = (lane >> 4) * 8;

    float cstate = 0.0f;                  // cell state for (b = w, j = lane)
    for (int step = 0; step < NT; ++step) {
        const int p = step & 1;
        const unsigned short* hprev = hb + p * 16 * H;
        unsigned short* hnext = hb + (1 - p) * 16 * H;

        v8f acc = {};
        for (int kt = 0; kt < 16; ++kt) {
            const int ks = kt * 32;
            __builtin_prefetch((const void*)(Bbase + ks + 32), 0, 0);
            Frag16 A, B;
            const unsigned short* hp = hprev + mrow * H + ks + koA;
            A.u[0] = *(const U4*)(hp);
            A.u[1] = *(const U4*)(hp + 16);
            B.u[0] = *(const U4*)(Bbase + ks);
            B.u[1] = *(const U4*)(Bbase + ks + 16);
            acc = __builtin_amdgcn_wmma_f32_16x16x32_bf16(
                false, A.v, false, B.v, (short)0, acc, false, false);
        }
        if (lane < 16) {
#pragma unroll
            for (int v = 0; v < 8; ++v)   // rows M=0..7 are b=0..7 (lanes<16)
                lds_g[sec][half * 16 + lane][v] = acc[v];
        }
        __syncthreads();

        // cell update: thread (w,lane) -> b = w (0..7), j = lane (0..31)
        const int t = (dd == 0) ? step : (NT - 1 - step);
        const int b = w, j = lane, col = slice0 + j;
        const float* ar = apre + (((size_t)rec * NT + t) * BS + b) * G + col;
        float gi = lds_g[0][j][b] + ar[0];
        float gf = lds_g[1][j][b] + ar[512];
        float gg = lds_g[2][j][b] + ar[1024];
        float go = lds_g[3][j][b] + ar[1536];
        cstate = sigm(gf) * cstate + sigm(gi) * tanhf(gg);
        float hv = sigm(go) * tanhf(cstate);
        hnext[b * H + col] = f2bf(hv);
        out[(((size_t)b * NT + t) * (2 * H) + dd * H + col) * KH + kk] = hv;

        rec_barrier(bar, (unsigned)(step + 1));   // also re-arms LDS for next step
    }
}

// ---------------- host launcher ----------------
extern "C" void kernel_launch(void* const* d_in, const int* in_sizes, int n_in,
                              void* d_out, int out_size, void* d_ws, size_t ws_size,
                              hipStream_t stream) {
    (void)in_sizes; (void)n_in; (void)out_size; (void)ws_size;
    const float* x     = (const float*)d_in[0];
    const float* Wih_f = (const float*)d_in[1];
    const float* Whh_f = (const float*)d_in[2];
    const float* bih_f = (const float*)d_in[3];
    const float* bhh_f = (const float*)d_in[4];
    const float* Wih_b = (const float*)d_in[5];
    const float* Whh_b = (const float*)d_in[6];
    const float* bih_b = (const float*)d_in[7];
    const float* bhh_b = (const float*)d_in[8];

    char* ws = (char*)d_ws;
    unsigned short* xbf  = (unsigned short*)(ws + OFF_X);
    unsigned short* wih  = (unsigned short*)(ws + OFF_WIH);
    unsigned short* whh  = (unsigned short*)(ws + OFF_WHH);
    float*          bias = (float*)(ws + OFF_BIAS);
    float*          apre = (float*)(ws + OFF_APRE);
    unsigned short* hbuf = (unsigned short*)(ws + OFF_HBUF);
    unsigned*       bars = (unsigned*)(ws + OFF_BARS);

    const size_t nWihD = (size_t)KH * G * CH;   // per-direction Wih elems
    const size_t nWhhD = (size_t)KH * G * H;    // per-direction Whh elems

    k_cvt<<<4096, 256, 0, stream>>>(Wih_f, wih, nWihD);
    k_cvt<<<4096, 256, 0, stream>>>(Wih_b, wih + nWihD, nWihD);
    k_cvt<<<4096, 256, 0, stream>>>(Whh_f, whh, nWhhD);
    k_cvt<<<4096, 256, 0, stream>>>(Whh_b, whh + nWhhD, nWhhD);
    k_cvt_x<<<4096, 256, 0, stream>>>(x, xbf);
    k_prep<<<512, 256, 0, stream>>>(bih_f, bhh_f, bih_b, bhh_b, bias, hbuf, bars);

    dim3 gg(G / 64, NTB / 64, NREC);            // 32 x 16 x 8
    k_gemm_in<<<gg, 128, 0, stream>>>(xbf, wih, bias, apre);

    k_lstm_rec<<<NREC * WGR, 256, 0, stream>>>(whh, apre, hbuf, (float*)d_out, bars);
}
